// LSTM_87067577025240
// MI455X (gfx1250) — matrix operand order repaired and checked by reading, not compiled, not run on hardware
//
#include <hip/hip_runtime.h>

typedef __attribute__((ext_vector_type(16))) __bf16 v16bf;
typedef __attribute__((ext_vector_type(8)))  float  v8f;

#define HID   512
#define SEQL  256
#define BATCH 2048
#define NCLS  10

typedef union { v16bf v; uint4 u[2]; } Frag;

// ---------------------------------------------------------------------------
// Prologue: pack fp32 recurrent weights into bf16, pre-swizzled into the
// V_WMMA_F32_16X16X32_BF16 A-fragment lane layout.
// Packed index: idx = (((g*32+mt)*16+kt)*32 + lane)*16 + j
//   lane 0-15 : M = mt*16 + lane,     elems 0-7 -> K=kt*32+0..7,  8-15 -> K=+16..23
//   lane 16-31: M = mt*16 + lane-16,  elems 0-7 -> K=kt*32+8..15, 8-15 -> K=+24..31
// Also packs bias[4*H] and w_x[4*H] as separate flat arrays.
// ---------------------------------------------------------------------------
__global__ __launch_bounds__(256) void lstm_prep(
    const float* __restrict__ wgh, const float* __restrict__ wih,
    const float* __restrict__ wfh, const float* __restrict__ woh,
    const float* __restrict__ wgx, const float* __restrict__ wix,
    const float* __restrict__ wfx, const float* __restrict__ wox,
    const float* __restrict__ bg,  const float* __restrict__ bi,
    const float* __restrict__ bf,  const float* __restrict__ bo,
    __bf16* __restrict__ wpack, float* __restrict__ bias_o,
    float* __restrict__ wx_o)
{
    int idx = blockIdx.x * 256 + threadIdx.x;           // 0 .. 4*512*512-1
    {
        int j    = idx & 15;
        int lane = (idx >> 4) & 31;
        int kt   = (idx >> 9) & 15;
        int mt   = (idx >> 13) & 31;
        int g    = (idx >> 18) & 3;
        int m = mt * 16 + (lane & 15);
        int k = kt * 32 + ((lane >> 4) << 3) + (j < 8 ? j : j + 8);
        const float* W = (g == 0) ? wgh : (g == 1) ? wih : (g == 2) ? wfh : woh;
        wpack[idx] = (__bf16)W[m * HID + k];
    }
    if (idx < 4 * HID) {
        int g = idx >> 9, m = idx & (HID - 1);
        const float* bb = (g == 0) ? bg  : (g == 1) ? bi  : (g == 2) ? bf  : bo;
        const float* wx = (g == 0) ? wgx : (g == 1) ? wix : (g == 2) ? wfx : wox;
        bias_o[idx] = bb[m];
        wx_o[idx]   = wx[m];
    }
}

// Load 4 A-fragments (2 gates x 2 M-tiles) for (kt, half) of this wave.
__device__ __forceinline__ void loadA4(Frag* buf, const uint4* wp4,
                                       int wave, int lane, int kt, int half)
{
#pragma unroll
    for (int gg = 0; gg < 2; ++gg)
#pragma unroll
        for (int mt = 0; mt < 2; ++mt) {
            int g = half * 2 + gg;
            int fragi = (g * 32 + wave * 2 + mt) * 16 + kt;
            const uint4* ap = wp4 + fragi * 64 + lane * 2;
            buf[gg * 2 + mt].u[0] = ap[0];
            buf[gg * 2 + mt].u[1] = ap[1];
        }
}

// ---------------------------------------------------------------------------
// Main kernel: one workgroup per 16 batch columns. 16 waves; each wave owns
// 2 M-tiles x 4 gates. h (bf16, transposed) in LDS, c in registers in the
// WMMA C layout. 128 v_wmma_f32_16x16x32_bf16 per wave per time step.
// WMMA phase is pure loads+WMMA: chains start from an inline-zero C operand;
// bias + w_x*x_t are added in the post-barrier elementwise phase.
// A-fragments are ping-pong prefetched in 4-fragment half-groups; the last
// half-group prefetches (kt=0, half=0) of the NEXT time step so the loads
// overlap the barrier + gate math. `tzero`==0 defeats LICM of weight loads.
// ---------------------------------------------------------------------------
__global__ __launch_bounds__(512) void lstm_main(
    const float*  __restrict__ x,      // [B][SEQ]
    const __bf16* __restrict__ wpack,  // packed A fragments
    const float*  __restrict__ biasg,  // [4*H]
    const float*  __restrict__ wxg,    // [4*H]
    const float*  __restrict__ h_init, // [H]
    const float*  __restrict__ c_init, // [H]
    const float*  __restrict__ wph,    // [C][H]
    const float*  __restrict__ bp,     // [C]
    float*        __restrict__ out,    // [B][C]
    int tzero)                         // == 0 (runtime)
{
    // LDS carve (static, 49408 B):
    //   [0      , 16640) : hT    bf16[16][520] (padded row stride)
    //   [16640  , 33024) : xs    f32 [16][256]
    //   [33024  , 41216) : biasS f32 [4*512]
    //   [41216  , 49408) : wxS   f32 [4*512]
    //   [16640  , 49408) : hfin  f32 [16][512] (overlay; written only after an
    //                      extra barrier on the final step)
    __shared__ __align__(16) unsigned char smem[49408];
    __bf16* hT    = (__bf16*)smem;
    float*  xs    = (float*)(smem + 16640);
    float*  biasS = (float*)(smem + 33024);
    float*  wxS   = (float*)(smem + 41216);
    float*  hfin  = (float*)(smem + 16640);

    const int tid  = threadIdx.x;
    const int wave = tid >> 5;
    const int lane = tid & 31;
    const int n    = lane & 15;            // batch column within tile (N)
    const int m0   = (lane >> 4) << 3;     // 0 or 8 (C layout M offset)
    const int n0   = blockIdx.x * 16;      // global batch base

    // preload per-tile x columns: xs[n][t]
    for (int i = tid; i < 16 * SEQL; i += 512) {
        int nn = i >> 8, t = i & 255;
        xs[nn * SEQL + t] = x[(n0 + nn) * SEQL + t];
    }
    // preload bias / w_x
    for (int i = tid; i < 4 * HID; i += 512) {
        biasS[i] = biasg[i];
        wxS[i]   = wxg[i];
    }
    // init hT[n][m] = h_init[m] (broadcast across batch)
    for (int i = tid; i < 16 * HID; i += 512) {
        int nn = i >> 9, m = i & 511;
        hT[nn * 520 + m] = (__bf16)h_init[m];
    }
    // init c in C-layout registers
    v8f c_st[2];
#pragma unroll
    for (int mt = 0; mt < 2; ++mt)
#pragma unroll
        for (int e = 0; e < 8; ++e)
            c_st[mt][e] = c_init[wave * 32 + mt * 16 + m0 + e];

    __syncthreads();

    const v8f zro = {};   // inline-0 C operand for the first K-tile

    // Prime the A-fragment pipeline for t=0, kt=0, half=0.
    Frag F0[4], F1[4];
    loadA4(F0, (const uint4*)wpack, wave, lane, 0, 0);

#pragma unroll 1
    for (int t = 0; t < SEQL; ++t) {
        // t-dependent (but runtime-identical) weight bases: defeat LICM.
        const uint4* wp4  = (const uint4*)wpack + (size_t)(t * tzero);
        const uint4* wp4n = (const uint4*)wpack + (size_t)((t + 1) * tzero);

        v8f acc[4][2];
        union { v16bf v; uint4 u[2]; } bfr;

        // 32 half-steps: u -> (kt = u>>1, half = u&1); 4 WMMAs each.
#pragma unroll
        for (int u = 0; u < 32; ++u) {
            int kt = u >> 1, half = u & 1;
            Frag* cur = (u & 1) ? F1 : F0;
            Frag* nxt = (u & 1) ? F0 : F1;
            if (u < 31)
                loadA4(nxt, wp4, wave, lane, (u + 1) >> 1, (u + 1) & 1);
            else
                loadA4(nxt, wp4n, wave, lane, 0, 0);  // next t (overlaps
                                                      // barrier + gate math)
            if (half == 0) {
                const uint4* hp = (const uint4*)(hT + n * 520 + kt * 32 +
                                                 ((lane >> 4) << 3));
                bfr.u[0] = hp[0];
                bfr.u[1] = hp[2];
            }
#pragma unroll
            for (int gg = 0; gg < 2; ++gg)
#pragma unroll
                for (int mt = 0; mt < 2; ++mt) {
                    int g = half * 2 + gg;
                    acc[g][mt] = __builtin_amdgcn_wmma_f32_16x16x32_bf16(
                        false, cur[gg * 2 + mt].v, false, bfr.v, (short)0,
                        (kt == 0) ? zro : acc[g][mt], false, false);
                }
        }
        __syncthreads();   // all hT reads done before rewrite

        // elementwise: add bias + w_x*x_t, gate math, state update, write h
        float xv = xs[n * SEQL + t];
        float hv[2][8];
#pragma unroll
        for (int mt = 0; mt < 2; ++mt) {
            int mbase = wave * 32 + mt * 16 + m0;
            union { uint4 u; __bf16 h8[8]; } hp;
#pragma unroll
            for (int e = 0; e < 8; ++e) {
                int m = mbase + e;
                float pg = acc[0][mt][e] + biasS[0 * HID + m] + wxS[0 * HID + m] * xv;
                float pi = acc[1][mt][e] + biasS[1 * HID + m] + wxS[1 * HID + m] * xv;
                float pf = acc[2][mt][e] + biasS[2 * HID + m] + wxS[2 * HID + m] * xv;
                float po = acc[3][mt][e] + biasS[3 * HID + m] + wxS[3 * HID + m] * xv;
                float gg = tanhf(pg);
                float ii = 1.f / (1.f + __expf(-pi));
                float ff = 1.f / (1.f + __expf(-pf));
                float oo = 1.f / (1.f + __expf(-po));
                float cc = gg * ii + c_st[mt][e] * ff;
                c_st[mt][e] = cc;
                float hh = tanhf(cc) * oo;
                hv[mt][e] = hh;
                hp.h8[e]  = (__bf16)hh;
            }
            *(uint4*)(hT + n * 520 + mbase) = hp.u;
        }
        if (t == SEQL - 1) {
            // hfin overlays biasS/wxS: wait for everyone's bias reads first.
            __syncthreads();
#pragma unroll
            for (int mt = 0; mt < 2; ++mt)
#pragma unroll
                for (int e = 0; e < 8; ++e)
                    hfin[n * HID + wave * 32 + mt * 16 + m0 + e] = hv[mt][e];
        }
        __syncthreads();   // h writes visible before next step's reads
    }

    // tiny fp32 projection: out[n0+n][c] = wph[c,:] . hfin[n,:] + bp[c]
    if (tid < 16 * NCLS) {
        int nn = tid / NCLS, c = tid % NCLS;
        float a = bp[c];
        const float* wr = wph + c * HID;
        const float* hr = hfin + nn * HID;
#pragma unroll 4
        for (int m = 0; m < HID; ++m) a += wr[m] * hr[m];
        out[(n0 + nn) * NCLS + c] = a;
    }
}

extern "C" void kernel_launch(void* const* d_in, const int* in_sizes, int n_in,
                              void* d_out, int out_size, void* d_ws, size_t ws_size,
                              hipStream_t stream) {
    (void)in_sizes; (void)n_in; (void)out_size; (void)ws_size;
    const float* x   = (const float*)d_in[0];
    const float* wgx = (const float*)d_in[1];
    const float* wix = (const float*)d_in[2];
    const float* wfx = (const float*)d_in[3];
    const float* wox = (const float*)d_in[4];
    const float* wgh = (const float*)d_in[5];
    const float* wih = (const float*)d_in[6];
    const float* wfh = (const float*)d_in[7];
    const float* woh = (const float*)d_in[8];
    const float* bg  = (const float*)d_in[9];
    const float* bi  = (const float*)d_in[10];
    const float* bf  = (const float*)d_in[11];
    const float* bo  = (const float*)d_in[12];
    const float* wph = (const float*)d_in[13];
    const float* bp  = (const float*)d_in[14];
    const float* hin = (const float*)d_in[15];
    const float* cin = (const float*)d_in[16];

    __bf16* wpack = (__bf16*)d_ws;
    float*  biasp = (float*)((char*)d_ws + (size_t)4 * HID * HID * sizeof(__bf16));
    float*  wxp   = biasp + 4 * HID;

    lstm_prep<<<4096, 256, 0, stream>>>(wgh, wih, wfh, woh,
                                        wgx, wix, wfx, wox,
                                        bg, bi, bf, bo, wpack, biasp, wxp);
    lstm_main<<<128, 512, 0, stream>>>(x, wpack, biasp, wxp, hin, cin, wph, bp,
                                       (float*)d_out, /*tzero=*/0);
}
